// Rudy_24017457119444
// MI455X (gfx1250) — matrix operand order repaired and checked by reading, not compiled, not run on hardware
//
#include <hip/hip_runtime.h>
#include <hip/hip_bf16.h>
#include <cstdint>

// ---------------------------------------------------------------------------
// RUDY congestion map as a bf16 WMMA GEMM:
//   h_map = ox^T @ (oy*wh) ,  v_map = ox^T @ (oy*wv)      (A panel shared!)
// Panels are precomputed bf16 [512 x K] (row-major, K fastest), K processed in
// 8 mega-chunks (<=12512 nets) so the 38 MB panel buffer is L2-resident.
// GEMM: 128x128 block/WG, wave = 128x16 strip -> 16 WMMA per wave per K-step.
// Panel tiles (3 x 128x32 bf16 = 24 KB/step) are DMA'd into double-buffered
// LDS by the Tensor Data Mover (tensor_load_to_lds + s_wait_tensorcnt),
// overlapping the WMMAs of the previous step.
// ---------------------------------------------------------------------------

namespace {
constexpr int   kNumNets   = 100000;
constexpr int   kNBX       = 512;
constexpr int   kNBY       = 512;
constexpr float kXL        = 0.0f;
constexpr float kYL        = 0.0f;
constexpr float kBSX       = 1000.0f / kNBX;   // 1.953125
constexpr float kBSY       = 1000.0f / kNBY;
constexpr float kCapH      = 0.1f;
constexpr float kCapV      = 0.1f;
constexpr int   kMega      = 8;                // K mega-chunks
constexpr int   kKcBase    = 12512;            // 32*391 (last chunk: 12416)
constexpr int   kKdwMax    = kKcBase / 2;      // panel row stride (dwords)
constexpr int   kKSplit    = 8;                // K partitions within a chunk
constexpr int   kBlock     = 128;              // 128x128 output block per WG
constexpr int   kTileDw    = kBlock * 16;      // dwords per staged panel tile
}

#if defined(__has_builtin)
#if __has_builtin(__builtin_amdgcn_tensor_load_to_lds)
#define USE_TDM 1
#endif
#endif
#ifndef USE_TDM
#define USE_TDM 0
#endif

typedef __attribute__((ext_vector_type(16))) __bf16 v16bf;
typedef __attribute__((ext_vector_type(8)))  float  v8f;
typedef __attribute__((ext_vector_type(4)))  unsigned vu32x4;
typedef __attribute__((ext_vector_type(8)))  int      vi32x8;
typedef __attribute__((ext_vector_type(4)))  int      vi32x4;

union Frag {
  uint4 q[2];
  v16bf v;
};

__device__ __forceinline__ unsigned pack2bf16(float a, float b) {
  // round-to-nearest-even fp32 -> bf16, packed (even K in [15:0], odd in [31:16])
  unsigned ua = __float_as_uint(a);
  unsigned ub = __float_as_uint(b);
  ua = (ua + 0x7FFFu + ((ua >> 16) & 1u)) >> 16;
  ub = (ub + 0x7FFFu + ((ub >> 16) & 1u)) >> 16;
  return ua | (ub << 16);
}

__device__ __forceinline__ float overlap1d(float lo, float hi, float elo, float bs) {
  float v = fminf(hi, elo + bs) - fmaxf(lo, elo);
  return fminf(fmaxf(v, 0.0f), bs);
}

#if USE_TDM
// ---------------------------------------------------------------------------
// TDM: DMA one 128-row x 32-element bf16 tile from a [kNBX x kElems] row-major
// tensor into LDS (compacted row-major, 64 B/row -> matches fragment layout).
// D# fields per CDNA5 ISA ch.8 (group0/1; groups 2-3 zero for a 2D tile).
// ---------------------------------------------------------------------------
__device__ __forceinline__ void tdm_load_tile(unsigned lds_addr,
                                              unsigned long long gaddr,
                                              int kElems) {
  vu32x4 g0 = {};
  g0[0] = 1u;                                        // count=1, user descriptor
  g0[1] = lds_addr;                                  // LDS byte address
  g0[2] = (unsigned)(gaddr & 0xffffffffull);         // global_addr[31:0]
  g0[3] = (unsigned)(gaddr >> 32) | (2u << 30);      // global_addr[56:32] | type=2
  vi32x8 g1 = {};
  g1[0] = 1 << 16;                                   // data_size=1 (2 bytes)
  g1[1] = (kElems & 0xffff) << 16;                   // tensor_dim0[15:0]
  g1[2] = ((kElems >> 16) & 0xffff) | (kNBX << 16);  // dim0[31:16] | tensor_dim1[15:0]
  g1[3] = 32 << 16;                                  // tile_dim0 = 32
  g1[4] = kBlock;                                    // tile_dim1 = 128, tile_dim2 = 0
  g1[5] = kElems;                                    // tensor_dim0_stride[31:0]
  g1[6] = 0;
  g1[7] = 0;
  vi32x4 z4 = {};
#if __clang_major__ >= 23
  vi32x8 z8 = {};
  __builtin_amdgcn_tensor_load_to_lds(g0, g1, z4, z4, z8, 0);
#else
  __builtin_amdgcn_tensor_load_to_lds(g0, g1, z4, z4, 0);
#endif
}

__device__ __forceinline__ void wait_tensorcnt0() {
#if __has_builtin(__builtin_amdgcn_s_wait_tensorcnt)
  __builtin_amdgcn_s_wait_tensorcnt(0);
#else
  asm volatile("s_wait_tensorcnt 0x0" ::: "memory");
#endif
}
#endif  // USE_TDM

// ---------------------------------------------------------------------------
// Kernel 0: zero the f32 accumulator maps (graph-replay deterministic).
// ---------------------------------------------------------------------------
__global__ void zero_kernel(float* __restrict__ p, int n) {
  int i = blockIdx.x * blockDim.x + threadIdx.x;
  if (i < n) p[i] = 0.0f;
}

// ---------------------------------------------------------------------------
// Kernel 1: per-net bounding boxes + scaled weights (SoA in workspace).
// ---------------------------------------------------------------------------
__global__ void bbox_kernel(const float* __restrict__ pin_pos,
                            const float* __restrict__ net_weights,
                            const int*   __restrict__ netpin_start,
                            const int*   __restrict__ flat_netpin,
                            int P, float* __restrict__ prm) {
  int n = blockIdx.x * blockDim.x + threadIdx.x;
  if (n >= kNumNets) return;
  int s = netpin_start[n];
  int e = netpin_start[n + 1];
  float xmn = 3.4e38f, xmx = -3.4e38f, ymn = 3.4e38f, ymx = -3.4e38f;
  for (int p = s; p < e; ++p) {
    int pid = flat_netpin[p];
    float x = pin_pos[pid];
    float y = pin_pos[P + pid];
    xmn = fminf(xmn, x); xmx = fmaxf(xmx, x);
    ymn = fminf(ymn, y); ymx = fmaxf(ymx, y);
  }
  float w  = net_weights[n];
  float dy = ymx - ymn, dx = xmx - xmn;
  prm[0 * kNumNets + n] = xmn;
  prm[1 * kNumNets + n] = xmx;
  prm[2 * kNumNets + n] = ymn;
  prm[3 * kNumNets + n] = ymx;
  prm[4 * kNumNets + n] = (dy > 0.0f) ? (w / dy) : 0.0f;  // wh
  prm[5 * kNumNets + n] = (dx > 0.0f) ? (w / dx) : 0.0f;  // wv
}

// ---------------------------------------------------------------------------
// Kernel 2: build bf16 panels for one K mega-chunk.
//   panel 0: ox[row][k]          (A, unscaled; shared by h and v)
//   panel 1: oy[row][k] * wh[k]  (B for h_map)
//   panel 2: oy[row][k] * wv[k]  (B for v_map)
// ---------------------------------------------------------------------------
__global__ void panel_kernel(const float* __restrict__ prm,
                             unsigned* __restrict__ panels,
                             int kg0, int kdw) {
  const int col = blockIdx.x * blockDim.x + threadIdx.x;   // packed-pair col
  if (col >= kdw) return;
  const int row   = blockIdx.y;                            // 0..511 bin
  const int panel = blockIdx.z;                            // 0..2
  const int g = kg0 + col * 2;                             // global net index
  unsigned out;
  if (panel == 0) {
    const float lo0 = prm[0 * kNumNets + g], lo1 = prm[0 * kNumNets + g + 1];
    const float hi0 = prm[1 * kNumNets + g], hi1 = prm[1 * kNumNets + g + 1];
    const float ex  = kXL + (float)row * kBSX;
    out = pack2bf16(overlap1d(lo0, hi0, ex, kBSX),
                    overlap1d(lo1, hi1, ex, kBSX));
  } else {
    const float lo0 = prm[2 * kNumNets + g], lo1 = prm[2 * kNumNets + g + 1];
    const float hi0 = prm[3 * kNumNets + g], hi1 = prm[3 * kNumNets + g + 1];
    const int warr  = (panel == 1) ? 4 : 5;
    const float w0  = prm[warr * kNumNets + g], w1 = prm[warr * kNumNets + g + 1];
    const float ey  = kYL + (float)row * kBSY;
    out = pack2bf16(overlap1d(lo0, hi0, ey, kBSY) * w0,
                    overlap1d(lo1, hi1, ey, kBSY) * w1);
  }
  panels[panel * (kNBX * kKdwMax) + row * kdw + col] = out;
}

// ---------------------------------------------------------------------------
// Shared WMMA consume step: fragment loads from staged tiles + 16 WMMAs.
// Fragment LDS addressing matches ISA 7.12.2 16-bit A(16x32)/B(32x16) layouts.
// ---------------------------------------------------------------------------
__device__ __forceinline__ void gemm_step(const unsigned* __restrict__ sA,
                                          const unsigned* __restrict__ sBH,
                                          const unsigned* __restrict__ sBV,
                                          int m, int half, int boff,
                                          v8f accH[8], v8f accV[8]) {
  Frag fBH, fBV;
  fBH.q[0] = *reinterpret_cast<const uint4*>(&sBH[boff]);
  fBH.q[1] = *reinterpret_cast<const uint4*>(&sBH[boff + 8]);
  fBV.q[0] = *reinterpret_cast<const uint4*>(&sBV[boff]);
  fBV.q[1] = *reinterpret_cast<const uint4*>(&sBV[boff + 8]);
#pragma unroll
  for (int xt = 0; xt < 8; ++xt) {
    const int aoff = (xt * 16 + m) * 16 + half * 4;
    Frag fA;
    fA.q[0] = *reinterpret_cast<const uint4*>(&sA[aoff]);
    fA.q[1] = *reinterpret_cast<const uint4*>(&sA[aoff + 8]);
    accH[xt] = __builtin_amdgcn_wmma_f32_16x16x32_bf16(
        false, fA.v, false, fBH.v, (short)0, accH[xt], false, false);
    accV[xt] = __builtin_amdgcn_wmma_f32_16x16x32_bf16(
        false, fA.v, false, fBV.v, (short)0, accV[xt], false, false);
  }
}

// ---------------------------------------------------------------------------
// Kernel 3: GEMM over one mega-chunk. 256 threads = 8 wave32, 128x128 block.
// ---------------------------------------------------------------------------
__global__ void __launch_bounds__(256)
rudy_wmma_kernel(const unsigned* __restrict__ pOX,
                 const unsigned* __restrict__ pBH,
                 const unsigned* __restrict__ pBV,
                 float* __restrict__ h_acc,
                 float* __restrict__ v_acc,
                 int kdw, int nsteps) {
  const int tid  = threadIdx.x;
  const int lane = tid & 31;
  const int wave = tid >> 5;

  const int BX = (blockIdx.x & 3) * kBlock;
  const int BY = (blockIdx.x >> 2) * kBlock;

  const int per = (nsteps + (int)gridDim.y - 1) / (int)gridDim.y;
  const int s0  = (int)blockIdx.y * per;
  int s1 = s0 + per; if (s1 > nsteps) s1 = nsteps;

  const int m    = lane & 15;
  const int half = lane >> 4;
  const int boff = (wave * 16 + m) * 16 + half * 4;

  v8f accH[8] = {};
  v8f accV[8] = {};

#if USE_TDM
  // Double-buffered LDS tiles: [buf][A | BH | BV], 24 KB per buffer.
  __shared__ unsigned sBuf[2][3 * kTileDw];
  const unsigned ldsBase = (unsigned)(uintptr_t)(&sBuf[0][0]);
  const int kElems = kdw * 2;

  const unsigned long long gA = (unsigned long long)(uintptr_t)pOX +
                                ((unsigned long long)(BX * kdw) << 2);
  const unsigned long long gH = (unsigned long long)(uintptr_t)pBH +
                                ((unsigned long long)(BY * kdw) << 2);
  const unsigned long long gV = (unsigned long long)(uintptr_t)pBV +
                                ((unsigned long long)(BY * kdw) << 2);

  if (wave == 0 && s0 < s1) {
    const unsigned long long off = (unsigned long long)(s0 * 16) << 2;
    tdm_load_tile(ldsBase,                     gA + off, kElems);
    tdm_load_tile(ldsBase + 4u * kTileDw,      gH + off, kElems);
    tdm_load_tile(ldsBase + 8u * kTileDw,      gV + off, kElems);
  }
  for (int s = s0; s < s1; ++s) {
    const int b = (s - s0) & 1;
    if (wave == 0) wait_tensorcnt0();
    __syncthreads();                        // LDS[b] tiles visible to all waves
    if (wave == 0 && s + 1 < s1) {
      const int bn = b ^ 1;
      const unsigned lb = ldsBase + (unsigned)(bn * 3 * kTileDw) * 4u;
      const unsigned long long off = (unsigned long long)((s + 1) * 16) << 2;
      tdm_load_tile(lb,                  gA + off, kElems);   // overlaps WMMAs
      tdm_load_tile(lb + 4u * kTileDw,   gH + off, kElems);
      tdm_load_tile(lb + 8u * kTileDw,   gV + off, kElems);
    }
    gemm_step(&sBuf[b][0], &sBuf[b][kTileDw], &sBuf[b][2 * kTileDw],
              m, half, boff, accH, accV);
    __syncthreads();                        // done reading before next overwrite
  }
#else
  // Fallback: per-thread staging (round-2 path).
  __shared__ unsigned sA [kTileDw];
  __shared__ unsigned sBH[kTileDw];
  __shared__ unsigned sBV[kTileDw];
  const int i0 = tid * 2, i1 = i0 + 1;
  const int r0 = i0 >> 2, c0 = (i0 & 3) * 4;
  const int r1 = i1 >> 2, c1 = (i1 & 3) * 4;
  for (int s = s0; s < s1; ++s) {
    const int kb = s * 16;
    *reinterpret_cast<uint4*>(&sA [i0 * 4]) =
        *reinterpret_cast<const uint4*>(&pOX[(BX + r0) * kdw + kb + c0]);
    *reinterpret_cast<uint4*>(&sA [i1 * 4]) =
        *reinterpret_cast<const uint4*>(&pOX[(BX + r1) * kdw + kb + c1]);
    *reinterpret_cast<uint4*>(&sBH[i0 * 4]) =
        *reinterpret_cast<const uint4*>(&pBH[(BY + r0) * kdw + kb + c0]);
    *reinterpret_cast<uint4*>(&sBH[i1 * 4]) =
        *reinterpret_cast<const uint4*>(&pBH[(BY + r1) * kdw + kb + c1]);
    *reinterpret_cast<uint4*>(&sBV[i0 * 4]) =
        *reinterpret_cast<const uint4*>(&pBV[(BY + r0) * kdw + kb + c0]);
    *reinterpret_cast<uint4*>(&sBV[i1 * 4]) =
        *reinterpret_cast<const uint4*>(&pBV[(BY + r1) * kdw + kb + c1]);
    __syncthreads();
    gemm_step(sA, sBH, sBV, m, half, boff, accH, accV);
    __syncthreads();
  }
#endif

  // C/D layout (32-bit 16x16): VGPR r: lanes<16 -> M=r, lanes>=16 -> M=r+8.
  const int ycol = BY + wave * 16 + m;
#pragma unroll
  for (int xt = 0; xt < 8; ++xt) {
    const int xb = BX + xt * 16 + half * 8;
#pragma unroll
    for (int r = 0; r < 8; ++r) {
      atomicAdd(&h_acc[(xb + r) * kNBY + ycol], accH[xt][r]);
      atomicAdd(&v_acc[(xb + r) * kNBY + ycol], accV[xt][r]);
    }
  }
}

// ---------------------------------------------------------------------------
// Kernel 4: out = max(|h|/(area*capH), |v|/(area*capV))
// ---------------------------------------------------------------------------
__global__ void finalize_kernel(const float* __restrict__ h_acc,
                                const float* __restrict__ v_acc,
                                float* __restrict__ out, int n) {
  int i = blockIdx.x * blockDim.x + threadIdx.x;
  if (i < n) {
    constexpr float sh = 1.0f / (kBSX * kBSY * kCapH);
    constexpr float sv = 1.0f / (kBSX * kBSY * kCapV);
    out[i] = fmaxf(fabsf(h_acc[i]) * sh, fabsf(v_acc[i]) * sv);
  }
}

extern "C" void kernel_launch(void* const* d_in, const int* in_sizes, int n_in,
                              void* d_out, int out_size, void* d_ws, size_t ws_size,
                              hipStream_t stream) {
  const float* pin_pos      = (const float*)d_in[0];
  const float* net_weights  = (const float*)d_in[1];
  const int*   netpin_start = (const int*)d_in[2];
  const int*   flat_netpin  = (const int*)d_in[3];
  const int P = in_sizes[0] / 2;

  // Workspace layout: prm (2.4 MB) | h_acc | v_acc (2 MB) | panels (38.4 MB)
  float*    prm    = (float*)d_ws;
  float*    h_acc  = prm + 6 * kNumNets;
  float*    v_acc  = h_acc + kNBX * kNBY;
  unsigned* panels = (unsigned*)(v_acc + kNBX * kNBY);
  unsigned* pOX = panels;
  unsigned* pBH = panels + 1 * (kNBX * kKdwMax);
  unsigned* pBV = panels + 2 * (kNBX * kKdwMax);

  const int nz = 2 * kNBX * kNBY;
  zero_kernel<<<(nz + 255) / 256, 256, 0, stream>>>(h_acc, nz);

  bbox_kernel<<<(kNumNets + 255) / 256, 256, 0, stream>>>(
      pin_pos, net_weights, netpin_start, flat_netpin, P, prm);

  int kg0 = 0;
  for (int mc = 0; mc < kMega; ++mc) {
    const int Kc     = (mc < kMega - 1) ? kKcBase : (kNumNets - (kMega - 1) * kKcBase);
    const int kdw    = Kc / 2;     // 6256 or 6208
    const int nsteps = Kc / 32;    // 391 or 388

    dim3 pg((kdw + 255) / 256, kNBX, 3);
    panel_kernel<<<pg, 256, 0, stream>>>(prm, panels, kg0, kdw);

    dim3 gg((kNBX / kBlock) * (kNBY / kBlock), kKSplit);   // 16 x 8
    rudy_wmma_kernel<<<gg, 256, 0, stream>>>(pOX, pBH, pBV, h_acc, v_acc,
                                             kdw, nsteps);
    kg0 += Kc;
  }

  const int n = kNBX * kNBY;
  finalize_kernel<<<(n + 255) / 256, 256, 0, stream>>>(h_acc, v_acc,
                                                       (float*)d_out, n);
}
